// MultiHeadAttention_47425028883105
// MI455X (gfx1250) — matrix operand order repaired and checked by reading, not compiled, not run on hardware
//
#include <hip/hip_runtime.h>

#define D_MODEL 1024
#define NHEADS  16
#define HDIM    64
#define BATCH   2
#define SEQ     2048
#define NROWS   (BATCH*SEQ)   // 4096
#define QKV_N   (3*D_MODEL)   // 3072

typedef __attribute__((ext_vector_type(16))) __bf16 bf16x16;
typedef __attribute__((ext_vector_type(8)))  float  f32x8;
typedef __attribute__((__vector_size__(16))) int    i32x4;

#if __has_builtin(__builtin_amdgcn_global_load_async_to_lds_b128) && \
    __has_builtin(__builtin_amdgcn_s_wait_asynccnt)
#define HAS_ASYNC_LDS 1
#endif

#define ASG __attribute__((address_space(1)))
#define ASL __attribute__((address_space(3)))

union FragU { bf16x16 v; uint4 q[2]; unsigned short u[16]; };

__device__ __forceinline__ unsigned short f2bf(float x){
  union { float f; unsigned int i; } c; c.f = x;
  unsigned int u = c.i;
  u += 0x7FFFu + ((u >> 16) & 1u);      // round-to-nearest-even
  return (unsigned short)(u >> 16);
}

// A-frag (16x32 bf16): lane holds row lane%16; halves 0-7 = K base..base+7,
// halves 8-15 = K base+16..base+23, base=(lane>>4)*8. caller passes row*lda+k+base.
__device__ __forceinline__ bf16x16 load_a_frag(const unsigned short* base){
  FragU f;
  f.q[0] = *(const uint4*)(base);
  f.q[1] = *(const uint4*)(base + 16);
  return f.v;
}
// B-frag (32x16 bf16): lane holds column lane%16; elems 0-7 = K base..base+7,
// elems 8-15 = K base+16..base+23, base=(lane>>4)*8. caller passes (k+base)*ld+col.
__device__ __forceinline__ bf16x16 load_b_frag(const unsigned short* p, int ld){
  FragU f;
  #pragma unroll
  for (int s = 0; s < 8; ++s){ f.u[s] = p[s*ld]; f.u[s+8] = p[(s+16)*ld]; }
  return f.v;
}

__device__ __forceinline__ f32x8 wmma_bf16(bf16x16 a, bf16x16 b, f32x8 c){
  return __builtin_amdgcn_wmma_f32_16x16x32_bf16(false, a, false, b, (short)0, c, false, false);
}

// 16B global -> LDS copy: async (ASYNCcnt-tracked) when available.
__device__ __forceinline__ void cp16_g2l(const unsigned short* g, unsigned short* l){
#ifdef HAS_ASYNC_LDS
  __builtin_amdgcn_global_load_async_to_lds_b128((ASG i32x4*)(g), (ASL i32x4*)(l), 0, 0);
#else
  *(uint4*)l = *(const uint4*)g;
#endif
}
__device__ __forceinline__ void async_wait0(){
#ifdef HAS_ASYNC_LDS
  __builtin_amdgcn_s_wait_asynccnt(0);
#endif
}
__device__ __forceinline__ void async_wait2(){
#ifdef HAS_ASYNC_LDS
  __builtin_amdgcn_s_wait_asynccnt(2);
#endif
}

// ---------------------------------------------------------------- fp32 -> bf16
__global__ void cvt_f32_bf16(const float* __restrict__ src,
                             unsigned short* __restrict__ dst, int n){
  int i = blockIdx.x*blockDim.x + threadIdx.x;
  int stride = gridDim.x*blockDim.x;
  for (; i < n; i += stride) dst[i] = f2bf(src[i]);
}

// ------------------------------------------------------------- shared GEMM core
// Block = 128 threads / 4 waves; tile 128(M) x 64(N); wave = 32x64 (2 M-frags).
// B tile (32x64 bf16, 4KB) staged in LDS, double-buffered via async-to-LDS.
__device__ __forceinline__ void stage_b(const unsigned short* gsrc, int ldb,
                                        unsigned short* ldst, int tid){
  #pragma unroll
  for (int j = 0; j < 2; ++j){
    const int c   = tid + 128*j;        // 16B chunk id, 0..255
    const int row = c >> 3;             // 8 chunks per 64-col row
    const int col = (c & 7) * 8;        // in ushorts
    cp16_g2l(gsrc + row*ldb + col, ldst + row*64 + col);
  }
}

template<int LDB>
__device__ __forceinline__ void gemm_mainloop(const unsigned short* __restrict__ A,
                                              const unsigned short* __restrict__ Bm,
                                              unsigned short* Bs,      // [2][32*64]
                                              int mblk, int n0, int tid,
                                              f32x8 (&acc)[2][4]){
  const int lane = tid & 31;
  const int wave = tid >> 5;
  const int koff = (lane >> 4) * 8;
  const int mrow = mblk + wave*32 + (lane & 15);

  stage_b(Bm + n0, LDB, Bs, tid);                       // k=0 -> buf0
  for (int k = 0; k < D_MODEL; k += 32){
    const int cur = (k >> 5) & 1;
    unsigned short* curb = Bs + cur*(32*64);
    if (k + 32 < D_MODEL){
      stage_b(Bm + (size_t)(k+32)*LDB + n0, LDB, Bs + (cur^1)*(32*64), tid);
      async_wait2();                                    // current tile landed
    } else {
      async_wait0();
    }
    __syncthreads();                                    // visible to all waves
    bf16x16 a0 = load_a_frag(A + (size_t)(mrow     )*D_MODEL + k + koff);
    bf16x16 a1 = load_a_frag(A + (size_t)(mrow + 16)*D_MODEL + k + koff);
    #pragma unroll
    for (int nf = 0; nf < 4; ++nf){
      bf16x16 b = load_b_frag(curb + koff*64 + nf*16 + (lane & 15), 64);
      acc[0][nf] = wmma_bf16(a0, b, acc[0][nf]);
      acc[1][nf] = wmma_bf16(a1, b, acc[1][nf]);
    }
    __syncthreads();                                    // done reading curb
  }
}

// ---------------------------------------------------------------- QKV GEMM
__global__ void __launch_bounds__(128)
qkv_gemm(const unsigned short* __restrict__ X,     // [4096,1024]
         const unsigned short* __restrict__ W,     // [1024,3072]
         const float* __restrict__ bias,
         unsigned short* __restrict__ Qb,          // [B,H,S,64]
         unsigned short* __restrict__ Kt,          // [B,H,64,S]
         unsigned short* __restrict__ Vb)          // [B,H,S,64]
{
  __shared__ __align__(16) unsigned short Bs[2*32*64];
  const int tid  = threadIdx.x;
  const int lane = tid & 31;
  const int wave = tid >> 5;
  const int mblk = blockIdx.x*128;
  const int n0   = blockIdx.y*64;

  f32x8 acc[2][4] = {};
  gemm_mainloop<QKV_N>(X, W, Bs, mblk, n0, tid, acc);

  #pragma unroll
  for (int mi = 0; mi < 2; ++mi){
    const int rbase = mblk + wave*32 + mi*16 + ((lane >> 4) ? 8 : 0);
    #pragma unroll
    for (int nf = 0; nf < 4; ++nf){
      const int col = n0 + nf*16 + (lane & 15);
      const float bv = bias[col];
      #pragma unroll
      for (int g = 0; g < 8; ++g){
        const int r = rbase + g;
        const int b = r >> 11, s = r & (SEQ-1);
        unsigned short val = f2bf(acc[mi][nf][g] + bv);
        if (col < D_MODEL){
          int h = col >> 6, d = col & 63;
          Qb[((b*NHEADS + h)*SEQ + s)*HDIM + d] = val;
        } else if (col < 2*D_MODEL){
          int c2 = col - D_MODEL, h = c2 >> 6, d = c2 & 63;
          Kt[((b*NHEADS + h)*HDIM + d)*SEQ + s] = val;   // K transposed
        } else {
          int c2 = col - 2*D_MODEL, h = c2 >> 6, d = c2 & 63;
          Vb[((b*NHEADS + h)*SEQ + s)*HDIM + d] = val;
        }
      }
    }
  }
}

// ---------------------------------------------------------------- flash attention
// Block = 64 query rows of one (b,h); 4 waves x 16 rows; key chunks of 128.
// K (64x128) and V (128x64) chunks staged in LDS via async-to-LDS.
__global__ void __launch_bounds__(128)
attn_kernel(const unsigned short* __restrict__ Qb,
            const unsigned short* __restrict__ Kt,
            const unsigned short* __restrict__ Vb,
            const int* __restrict__ mask,
            unsigned short* __restrict__ Ctx)          // [4096,1024] = [B,S,H*64]
{
  __shared__ __align__(16) unsigned short Ks[64*128];   // [d][key]
  __shared__ __align__(16) unsigned short Vs[128*64];   // [key][d]
  __shared__ __align__(16) unsigned short Pl[4*16*128]; // per-wave P tile
  const int tid  = threadIdx.x;
  const int lane = tid & 31;
  const int wave = tid >> 5;
  const int qt = blockIdx.x;            // 0..31
  const int bh = blockIdx.y;            // 0..31
  const int b  = bh >> 4;
  const int h  = bh & 15;
  const int q0 = qt*64 + wave*16;
  const int koff = (lane >> 4) * 8;
  const int ln15 = lane & 15;

  const unsigned short* qp = Qb + ((size_t)bh*SEQ + q0 + ln15)*HDIM;
  bf16x16 aq0 = load_a_frag(qp + 0  + koff);
  bf16x16 aq1 = load_a_frag(qp + 32 + koff);

  f32x8 acc[4] = {};
  float m[8], l[8];
  #pragma unroll
  for (int g = 0; g < 8; ++g){ m[g] = -1e30f; l[g] = 0.f; }

  unsigned short* pw = Pl + wave*16*128;
  const unsigned short* kbase = Kt + (size_t)bh*HDIM*SEQ;
  const unsigned short* vbase = Vb + (size_t)bh*SEQ*HDIM;

  for (int kc = 0; kc < SEQ; kc += 128){
    // ---- stage K/V chunk into LDS (prev chunk's reads finished at trailing barrier)
    __syncthreads();
    #pragma unroll
    for (int j = 0; j < 8; ++j){                       // K: 64 x 128 = 16KB
      const int c   = tid + 128*j;                     // 16B chunks, 0..1023
      const int row = c >> 4, col = (c & 15) * 8;
      cp16_g2l(kbase + (size_t)row*SEQ + kc + col, Ks + row*128 + col);
    }
    {
      const unsigned short* vsrc = vbase + (size_t)kc*HDIM;   // contiguous 16KB
      #pragma unroll
      for (int j = 0; j < 8; ++j){
        const int off = (tid + 128*j) * 8;
        cp16_g2l(vsrc + off, Vs + off);
      }
    }
    async_wait0();
    __syncthreads();

    // ---- scores S = (Q K^T) * 1/sqrt(64), masked   (K-frags from LDS)
    f32x8 sf[8];
    #pragma unroll
    for (int nf = 0; nf < 8; ++nf){
      f32x8 s = {};
      s = wmma_bf16(aq0, load_b_frag(Ks + (     koff)*128 + nf*16 + ln15, 128), s);
      s = wmma_bf16(aq1, load_b_frag(Ks + (32 + koff)*128 + nf*16 + ln15, 128), s);
      const int col = kc + nf*16 + ln15;
      const float mk = (mask[b*SEQ + col] != 0) ? 0.f : -1e30f;
      #pragma unroll
      for (int g = 0; g < 8; ++g) s[g] = s[g]*0.125f + mk;
      sf[nf] = s;
    }
    // ---- online softmax: row max across 16-lane half + 8 N-frags, rescale
    #pragma unroll
    for (int g = 0; g < 8; ++g){
      float mloc = -1e30f;
      #pragma unroll
      for (int nf = 0; nf < 8; ++nf) mloc = fmaxf(mloc, sf[nf][g]);
      #pragma unroll
      for (int msk = 1; msk < 16; msk <<= 1) mloc = fmaxf(mloc, __shfl_xor(mloc, msk, 32));
      const float mnew = fmaxf(m[g], mloc);
      const float corr = __expf(m[g] - mnew);
      m[g] = mnew;
      l[g] *= corr;
      #pragma unroll
      for (int nf = 0; nf < 4; ++nf) acc[nf][g] *= corr;
    }
    // ---- P = exp(S - m) -> LDS (C-layout -> row-major), accumulate row sums
    __syncthreads();
    const int rowhalf = (lane >> 4) * 8;
    #pragma unroll
    for (int g = 0; g < 8; ++g){
      float ls = 0.f;
      #pragma unroll
      for (int nf = 0; nf < 8; ++nf){
        float p = __expf(sf[nf][g] - m[g]);
        ls += p;
        pw[(g + rowhalf)*128 + nf*16 + ln15] = f2bf(p);
      }
      #pragma unroll
      for (int msk = 1; msk < 16; msk <<= 1) ls += __shfl_xor(ls, msk, 32);
      l[g] += ls;
    }
    __syncthreads();
    // ---- ctx += P @ V  (P A-frags and V B-frags from LDS)
    #pragma unroll
    for (int ks = 0; ks < 4; ++ks){
      bf16x16 ap = load_a_frag(pw + ln15*128 + ks*32 + koff);
      #pragma unroll
      for (int nf = 0; nf < 4; ++nf)
        acc[nf] = wmma_bf16(ap, load_b_frag(Vs + (ks*32 + koff)*64 + nf*16 + ln15, 64), acc[nf]);
    }
  }
  // ---- normalize, write ctx (bf16) in [B,S,H*hd] layout
  const int rbase = q0 + ((lane >> 4) ? 8 : 0);
  #pragma unroll
  for (int nf = 0; nf < 4; ++nf){
    #pragma unroll
    for (int g = 0; g < 8; ++g){
      const int srow = rbase + g;
      Ctx[((size_t)b*SEQ + srow)*D_MODEL + h*HDIM + nf*16 + ln15] = f2bf(acc[nf][g] / l[g]);
    }
  }
}

// ---------------------------------------------------------------- output projection
__global__ void __launch_bounds__(128)
proj_gemm(const unsigned short* __restrict__ Ctx,
          const unsigned short* __restrict__ W,
          const float* __restrict__ bias,
          float* __restrict__ Out)
{
  __shared__ __align__(16) unsigned short Bs[2*32*64];
  const int tid  = threadIdx.x;
  const int lane = tid & 31;
  const int wave = tid >> 5;
  const int mblk = blockIdx.x*128;
  const int n0   = blockIdx.y*64;

  f32x8 acc[2][4] = {};
  gemm_mainloop<D_MODEL>(Ctx, W, Bs, mblk, n0, tid, acc);

  #pragma unroll
  for (int mi = 0; mi < 2; ++mi){
    const int rbase = mblk + wave*32 + mi*16 + ((lane >> 4) ? 8 : 0);
    #pragma unroll
    for (int nf = 0; nf < 4; ++nf){
      const int col = n0 + nf*16 + (lane & 15);
      const float bv = bias[col];
      #pragma unroll
      for (int g = 0; g < 8; ++g)
        Out[(size_t)(rbase + g)*D_MODEL + col] = acc[mi][nf][g] + bv;
    }
  }
}

// ----------------------------------------------------------------
extern "C" void kernel_launch(void* const* d_in, const int* in_sizes, int n_in,
                              void* d_out, int out_size, void* d_ws, size_t ws_size,
                              hipStream_t stream){
  const float* x     = (const float*)d_in[0];
  const int*   mask  = (const int*)  d_in[1];
  const float* Wqkv  = (const float*)d_in[2];
  const float* bqkv  = (const float*)d_in[3];
  const float* Wproj = (const float*)d_in[4];
  const float* bproj = (const float*)d_in[5];
  float* out = (float*)d_out;

  char* ws = (char*)d_ws;
  size_t o = 0;
  unsigned short* xb     = (unsigned short*)(ws + o); o += (size_t)NROWS*D_MODEL*2;
  unsigned short* wqkvb  = (unsigned short*)(ws + o); o += (size_t)D_MODEL*QKV_N*2;
  unsigned short* wprojb = (unsigned short*)(ws + o); o += (size_t)D_MODEL*D_MODEL*2;
  unsigned short* Qb     = (unsigned short*)(ws + o); o += (size_t)NROWS*D_MODEL*2;
  unsigned short* Kt     = (unsigned short*)(ws + o); o += (size_t)NROWS*D_MODEL*2;
  unsigned short* Vb     = (unsigned short*)(ws + o); o += (size_t)NROWS*D_MODEL*2;
  unsigned short* Ctx    = (unsigned short*)(ws + o); o += (size_t)NROWS*D_MODEL*2;

  cvt_f32_bf16<<<512, 256, 0, stream>>>(x,     xb,     NROWS*D_MODEL);
  cvt_f32_bf16<<<512, 256, 0, stream>>>(Wqkv,  wqkvb,  D_MODEL*QKV_N);
  cvt_f32_bf16<<<512, 256, 0, stream>>>(Wproj, wprojb, D_MODEL*D_MODEL);

  qkv_gemm<<<dim3(NROWS/128, QKV_N/64), 128, 0, stream>>>(xb, wqkvb, bqkv, Qb, Kt, Vb);
  attn_kernel<<<dim3(SEQ/64, BATCH*NHEADS), 128, 0, stream>>>(Qb, Kt, Vb, mask, Ctx);
  proj_gemm<<<dim3(NROWS/128, D_MODEL/64), 128, 0, stream>>>(Ctx, wprojb, bproj, out);
}
